// AttnDecoderRNN_16655883173970
// MI455X (gfx1250) — matrix operand order repaired and checked by reading, not compiled, not run on hardware
//
#include <hip/hip_runtime.h>

// ---------------------------------------------------------------------------
// AttnDecoderRNN on MI455X (gfx1250): bf16 WMMA GEMMs + fused attention/GRU.
// B=64, S=64, H=1024, V=32000, T=40.
// ---------------------------------------------------------------------------

#define B_  64
#define S_  64
#define H_  1024
#define V_  32000
#define T_  40

typedef __attribute__((ext_vector_type(16))) __bf16 v16bf;
typedef __attribute__((ext_vector_type(8)))  float  v8f;
typedef int v4i_vs __attribute__((vector_size(16)));   // matches builtin param

struct bfx16 { uint4 lo; uint4 hi; };   // 32 bytes == v16bf

__device__ __forceinline__ unsigned short f2bf(float f) {
    unsigned u = __builtin_bit_cast(unsigned, f);
    unsigned r = u + 0x7FFFu + ((u >> 16) & 1u);   // round-to-nearest-even
    return (unsigned short)(r >> 16);
}

__device__ __forceinline__ float wave_sum(float v) {
    #pragma unroll
    for (int off = 16; off > 0; off >>= 1) v += __shfl_xor(v, off, 32);
    return v;
}
__device__ __forceinline__ float wave_max(float v) {
    #pragma unroll
    for (int off = 16; off > 0; off >>= 1) v = fmaxf(v, __shfl_xor(v, off, 32));
    return v;
}

// ---------------------------------------------------------------------------
// Async global->LDS staging (CDNA5 GLOBAL_LOAD_ASYNC_TO_LDS_B128, ASYNCcnt).
// Probe round 2 revealed param 1 is 'vector_size(16) int __device__ *', so we
// pass exact-pointee-type pointers and let clang's builtin AS-rewrite handle
// the language address spaces. Guarded: falls back to VGPR round-trip copy.
// ---------------------------------------------------------------------------
#if defined(__AMDGCN__) && __has_builtin(__builtin_amdgcn_global_load_async_to_lds_b128)
#define HAVE_ASYNC_LDS 1
__device__ __forceinline__ void async_g2l_b128(const void* g, void* l) {
    __builtin_amdgcn_global_load_async_to_lds_b128(
        (v4i_vs*)g, (v4i_vs*)l, 0, 0);
}
__device__ __forceinline__ void wait_async_lds() {
#if __has_builtin(__builtin_amdgcn_s_wait_asynccnt)
    __builtin_amdgcn_s_wait_asynccnt(0);
#else
    asm volatile("s_wait_asynccnt 0x0" ::: "memory");
#endif
}
#else
#define HAVE_ASYNC_LDS 0
#endif

// ---------------------------------------------------------------------------
// Generic GEMM: C[M,N] = A[M,K](bf16) * W[N,K](bf16)^T + bias[N]  (fp32 out)
// Block = 256 threads (8 waves). Block tile: 64 rows x 128 cols.
// grid.x = N/128, grid.y = M/64. K multiple of 512. LDS stages 64xKB A panel.
// Each wave: one 16-col tile, all four 16-row tiles (reuses its B fragment 4x).
// ---------------------------------------------------------------------------
#define KB_ 512

__global__ void __launch_bounds__(256)
gemm_bf16_bt(const unsigned short* __restrict__ A16,
             const unsigned short* __restrict__ W16,
             const float* __restrict__ bias,
             float* __restrict__ C, int N, int K)
{
    const __bf16* A = (const __bf16*)A16;
    const __bf16* W = (const __bf16*)W16;

    __shared__ __bf16 As[64 * KB_];          // 64 KB

    const int tid  = threadIdx.x;
    const int lane = tid & 31;
    const int wave = tid >> 5;
    const int ntile = blockIdx.x * 128 + wave * 16;
    const int col   = ntile + (lane & 15);
    const int lo    = (lane < 16) ? 0 : 8;    // A-frag K sub-offset
    const int khalf = (lane < 16) ? 0 : 8;    // B-frag K sub-offset

    const __bf16* Apanel = A + (size_t)blockIdx.y * 64 * K;
    const __bf16* Wcol   = W + (size_t)col * K + khalf;

    v8f acc[4] = {};

    for (int kb = 0; kb < K; kb += KB_) {
        // ---- stage A[panel, kb:kb+KB_] into LDS (16B = 8 halves per op)
#if HAVE_ASYNC_LDS
        for (int i = tid; i < 64 * KB_ / 8; i += 256) {
            int row = i >> 6;                 // KB_/8 = 64 chunks per row
            int c8  = (i & 63) << 3;          // half offset within row
            async_g2l_b128(Apanel + (size_t)row * K + kb + c8,
                           As + row * KB_ + c8);
        }
        wait_async_lds();
#else
        for (int i = tid; i < 64 * KB_ / 8; i += 256) {
            int row = i >> 6;
            int c8  = (i & 63) << 3;
            *(uint4*)(As + row * KB_ + c8) =
                *(const uint4*)(Apanel + (size_t)row * K + kb + c8);
        }
#endif
        __syncthreads();

        for (int k = 0; k < KB_; k += 32) {
            // B fragment: two 8-bf16 chunks (K = k+khalf.., k+16+khalf..) from this wave's W row (L2-resident)
            bfx16 tb;
            tb.lo = *(const uint4*)(Wcol + kb + k);
            tb.hi = *(const uint4*)(Wcol + kb + k + 16);
            v16bf bfrag = __builtin_bit_cast(v16bf, tb);
            #pragma unroll
            for (int m = 0; m < 4; ++m) {
                const __bf16* ap = As + (m * 16 + (lane & 15)) * KB_ + k + lo;
                bfx16 t;
                t.lo = *(const uint4*)ap;         // K = k+lo   .. +7
                t.hi = *(const uint4*)(ap + 16);  // K = k+16+lo.. +7
                v16bf afrag = __builtin_bit_cast(v16bf, t);
                acc[m] = __builtin_amdgcn_wmma_f32_16x16x32_bf16(
                             false, afrag, false, bfrag,
                             (short)0, acc[m], false, false);
            }
        }
        __syncthreads();
    }

    const float bv = bias ? bias[col] : 0.0f;
    const int rsub = (lane < 16) ? 0 : 8;
    #pragma unroll
    for (int m = 0; m < 4; ++m) {
        int gm0 = blockIdx.y * 64 + m * 16 + rsub;
        #pragma unroll
        for (int j = 0; j < 8; ++j)
            C[(size_t)(gm0 + j) * N + col] = acc[m][j] + bv;
    }
}

// ---------------------------------------------------------------------------
// fp32 -> bf16 convert
// ---------------------------------------------------------------------------
__global__ void __launch_bounds__(256)
cvt_f32_bf16_k(const float* __restrict__ src, unsigned short* __restrict__ dst,
               long long n)
{
    long long i = (long long)blockIdx.x * 256 + threadIdx.x;
    long long stride = (long long)gridDim.x * 256;
    for (; i < n; i += stride) dst[i] = f2bf(src[i]);
}

// h0 = encoder_hidden (fp32 copy + bf16 shadow)
__global__ void __launch_bounds__(256)
init_h_k(const float* __restrict__ eh, float* __restrict__ h,
         unsigned short* __restrict__ h_bf)
{
    int i = blockIdx.x * 256 + threadIdx.x;   // grid covers B*H
    float v = eh[i];
    h[i] = v;
    h_bf[i] = f2bf(v);
}

// ---------------------------------------------------------------------------
// Fused attention step (one block per batch row):
//   scores[s] = Va . tanh(q[b] + keysU[b,s]) + bv ;  w = softmax(scores)
//   ctx = sum_s w[s] * enc[b,s] ;  x_bf = [embed(token_t) ; ctx] (bf16)
//   attn weights written straight to d_out attention region.
// ---------------------------------------------------------------------------
__global__ void __launch_bounds__(256)
attn_step_k(const float* __restrict__ q, const float* __restrict__ keysU,
            const float* __restrict__ Va, const float* __restrict__ bv,
            const float* __restrict__ enc, const float* __restrict__ emb,
            const int* __restrict__ target,
            unsigned short* __restrict__ x_bf,
            float* __restrict__ attn_out, int t)
{
    const int b = blockIdx.x;
    const int tid = threadIdx.x;
    const int lane = tid & 31, wave = tid >> 5;

    __shared__ float sc[S_];
    __shared__ float wsh[S_];

    const float* qb = q + (size_t)b * H_;

    // scores: each wave owns 8 consecutive s values
    #pragma unroll
    for (int i = 0; i < 8; ++i) {
        int s = wave * 8 + i;
        const float* ku = keysU + ((size_t)b * S_ + s) * H_;
        float p = 0.f;
        for (int h = lane; h < H_; h += 32)
            p += Va[h] * tanhf(qb[h] + ku[h]);
        p = wave_sum(p);
        if (lane == 0) sc[s] = p + bv[0];
    }
    __syncthreads();

    // softmax over S=64 (redundant per-thread reduction; tiny)
    float mx = -INFINITY;
    #pragma unroll
    for (int s = 0; s < S_; ++s) mx = fmaxf(mx, sc[s]);
    float sum = 0.f;
    #pragma unroll
    for (int s = 0; s < S_; ++s) sum += expf(sc[s] - mx);
    float inv = 1.0f / sum;
    if (tid < S_) {
        float w = expf(sc[tid] - mx) * inv;
        wsh[tid] = w;
        attn_out[((size_t)b * T_ + t) * S_ + tid] = w;
    }
    __syncthreads();

    // context + x assembly
    const int token = (t == 0) ? 0 : target[b * T_ + (t - 1)];
    const float* er = emb + (size_t)token * H_;
    const float* eb = enc + (size_t)b * S_ * H_;
    for (int h = tid; h < H_; h += 256) {
        x_bf[(size_t)b * (2 * H_) + h] = f2bf(er[h]);
        float c = 0.f;
        #pragma unroll 8
        for (int s = 0; s < S_; ++s) c += wsh[s] * eb[(size_t)s * H_ + h];
        x_bf[(size_t)b * (2 * H_) + H_ + h] = f2bf(c);
    }
}

// ---------------------------------------------------------------------------
// GRU gates: h <- (1-z)*n + z*h ; also refresh bf16 shadow of h.
// gi/gh are [B, 3H] = [r | z | n] pre-activations. grid covers B*H.
// ---------------------------------------------------------------------------
__global__ void __launch_bounds__(256)
gru_gates_k(const float* __restrict__ gi, const float* __restrict__ gh,
            float* __restrict__ h, unsigned short* __restrict__ h_bf)
{
    int idx = blockIdx.x * 256 + threadIdx.x;   // 0 .. B*H-1
    int b = idx >> 10, hh = idx & (H_ - 1);
    size_t base = (size_t)b * (3 * H_) + hh;
    float ir = gi[base], iz = gi[base + H_], in = gi[base + 2 * H_];
    float hr = gh[base], hz = gh[base + H_], hn = gh[base + 2 * H_];
    float r = 1.0f / (1.0f + expf(-(ir + hr)));
    float z = 1.0f / (1.0f + expf(-(iz + hz)));
    float n = tanhf(in + r * hn);
    float hv = (1.0f - z) * n + z * h[idx];
    h[idx] = hv;
    h_bf[idx] = f2bf(hv);
}

// ---------------------------------------------------------------------------
// Row-wise log-softmax over V=32000, writes d_out[b, t, :].
// ---------------------------------------------------------------------------
__global__ void __launch_bounds__(256)
logsoftmax_k(const float* __restrict__ logits, float* __restrict__ out, int t)
{
    const int b = blockIdx.x, tid = threadIdx.x;
    const float* row = logits + (size_t)b * V_;
    __shared__ float red[8];

    float mx = -INFINITY;
    for (int v = tid; v < V_; v += 256) mx = fmaxf(mx, row[v]);
    mx = wave_max(mx);
    if ((tid & 31) == 0) red[tid >> 5] = mx;
    __syncthreads();
    mx = red[0];
    #pragma unroll
    for (int i = 1; i < 8; ++i) mx = fmaxf(mx, red[i]);
    __syncthreads();

    float sum = 0.f;
    for (int v = tid; v < V_; v += 256) sum += expf(row[v] - mx);
    sum = wave_sum(sum);
    if ((tid & 31) == 0) red[tid >> 5] = sum;
    __syncthreads();
    sum = 0.f;
    #pragma unroll
    for (int i = 0; i < 8; ++i) sum += red[i];
    float lse = mx + logf(sum);

    float* o = out + ((size_t)b * T_ + t) * V_;
    for (int v = tid; v < V_; v += 256) o[v] = row[v] - lse;
}

__global__ void __launch_bounds__(256)
copy_f32_k(const float* __restrict__ src, float* __restrict__ dst)
{
    int i = blockIdx.x * 256 + threadIdx.x;
    dst[i] = src[i];
}

// ---------------------------------------------------------------------------
// Host driver
// ---------------------------------------------------------------------------
static inline void launch_gemm(const unsigned short* A, const unsigned short* W,
                               const float* bias, float* C,
                               int M, int N, int K, hipStream_t s)
{
    dim3 grid(N / 128, M / 64);
    gemm_bf16_bt<<<grid, 256, 0, s>>>(A, W, bias, C, N, K);
}

static inline void launch_cvt(const float* src, unsigned short* dst,
                              long long n, hipStream_t s)
{
    int blocks = (int)((n + 255) / 256);
    cvt_f32_bf16_k<<<blocks, 256, 0, s>>>(src, dst, n);
}

extern "C" void kernel_launch(void* const* d_in, const int* in_sizes, int n_in,
                              void* d_out, int out_size, void* d_ws, size_t ws_size,
                              hipStream_t stream)
{
    (void)in_sizes; (void)n_in; (void)out_size; (void)ws_size;

    const float* enc  = (const float*)d_in[0];   // [B,S,H]
    const float* eh   = (const float*)d_in[1];   // [B,H]
    const int*   tgt  = (const int*)  d_in[2];   // [B,T]
    const float* emb  = (const float*)d_in[3];   // [V,H]
    const float* Wa   = (const float*)d_in[4];   // [H,H]
    const float* ba   = (const float*)d_in[5];   // [H]
    const float* Ua   = (const float*)d_in[6];   // [H,H]
    const float* bu   = (const float*)d_in[7];   // [H]
    const float* Va   = (const float*)d_in[8];   // [1,H]
    const float* bv   = (const float*)d_in[9];   // [1]
    const float* Wih  = (const float*)d_in[10];  // [3H,2H]
    const float* Whh  = (const float*)d_in[11];  // [3H,H]
    const float* bih  = (const float*)d_in[12];  // [3H]
    const float* bhh  = (const float*)d_in[13];  // [3H]
    const float* Wout = (const float*)d_in[14];  // [V,H]
    const float* bout = (const float*)d_in[15];  // [V]
    float* out = (float*)d_out;

    // d_out layout: decoder_outputs [B,T,V] | decoder_hidden [B,H] | attn [B,T,S]
    float* out_logp   = out;
    float* out_hidden = out + (size_t)B_ * T_ * V_;
    float* out_attn   = out_hidden + (size_t)B_ * H_;

    // workspace bump allocator (256B aligned)
    char* ws = (char*)d_ws;
    size_t off = 0;
    auto alloc = [&](size_t bytes) -> void* {
        void* p = ws + off;
        off = (off + bytes + 255) & ~(size_t)255;
        return p;
    };

    unsigned short* Wa_b   = (unsigned short*)alloc((size_t)H_ * H_ * 2);
    unsigned short* Ua_b   = (unsigned short*)alloc((size_t)H_ * H_ * 2);
    unsigned short* Wih_b  = (unsigned short*)alloc((size_t)3 * H_ * 2 * H_ * 2);
    unsigned short* Whh_b  = (unsigned short*)alloc((size_t)3 * H_ * H_ * 2);
    unsigned short* Wout_b = (unsigned short*)alloc((size_t)V_ * H_ * 2);
    unsigned short* enc_b  = (unsigned short*)alloc((size_t)B_ * S_ * H_ * 2);
    float*          keysU  = (float*)alloc((size_t)B_ * S_ * H_ * 4);
    float*          h      = (float*)alloc((size_t)B_ * H_ * 4);
    unsigned short* h_b    = (unsigned short*)alloc((size_t)B_ * H_ * 2);
    float*          q      = (float*)alloc((size_t)B_ * H_ * 4);
    unsigned short* x_b    = (unsigned short*)alloc((size_t)B_ * 2 * H_ * 2);
    float*          gi     = (float*)alloc((size_t)B_ * 3 * H_ * 4);
    float*          gh     = (float*)alloc((size_t)B_ * 3 * H_ * 4);
    float*          logits = (float*)alloc((size_t)B_ * V_ * 4);

    // --- one-time conversions (re-done every call: deterministic, ~8us) ---
    launch_cvt(Wa,   Wa_b,   (long long)H_ * H_,          stream);
    launch_cvt(Ua,   Ua_b,   (long long)H_ * H_,          stream);
    launch_cvt(Wih,  Wih_b,  (long long)3 * H_ * 2 * H_,  stream);
    launch_cvt(Whh,  Whh_b,  (long long)3 * H_ * H_,      stream);
    launch_cvt(Wout, Wout_b, (long long)V_ * H_,          stream);
    launch_cvt(enc,  enc_b,  (long long)B_ * S_ * H_,     stream);

    // keysU = enc @ Ua.T + bu   (M=B*S=4096, N=H, K=H)
    launch_gemm(enc_b, Ua_b, bu, keysU, B_ * S_, H_, H_, stream);

    // h0 = encoder_hidden
    init_h_k<<<(B_ * H_) / 256, 256, 0, stream>>>(eh, h, h_b);

    // --- 40 sequential decode steps ---
    for (int t = 0; t < T_; ++t) {
        // q = h @ Wa.T + ba
        launch_gemm(h_b, Wa_b, ba, q, B_, H_, H_, stream);
        // attention + x assembly
        attn_step_k<<<B_, 256, 0, stream>>>(q, keysU, Va, bv, enc, emb, tgt,
                                            x_b, out_attn, t);
        // gi = [emb;ctx] @ W_ih.T + b_ih   (K = 2H)
        launch_gemm(x_b, Wih_b, bih, gi, B_, 3 * H_, 2 * H_, stream);
        // gh = h @ W_hh.T + b_hh
        launch_gemm(h_b, Whh_b, bhh, gh, B_, 3 * H_, H_, stream);
        // GRU update (h, h_b refreshed)
        gru_gates_k<<<(B_ * H_) / 256, 256, 0, stream>>>(gi, gh, h, h_b);
        // logits = h_new @ W_out.T + b_out  (W_out bf16 is L2-resident: 65.5 MB)
        launch_gemm(h_b, Wout_b, bout, logits, B_, V_, H_, stream);
        // log_softmax -> d_out[:, t, :]
        logsoftmax_k<<<B_, 256, 0, stream>>>(logits, out_logp, t);
    }

    // decoder_hidden = final h
    copy_f32_k<<<(B_ * H_) / 256, 256, 0, stream>>>(h, out_hidden);
}
